// RGBfield_cnn_3667902071315
// MI455X (gfx1250) — matrix operand order repaired and checked by reading, not compiled, not run on hardware
//
#include <hip/hip_runtime.h>

typedef __attribute__((ext_vector_type(16))) __bf16 v16bf;
typedef __attribute__((ext_vector_type(8)))  float  v8f;

union FragBF { uint4 u[2]; v16bf v; };

__device__ __forceinline__ unsigned short f2bf(float f) {
  unsigned u = __builtin_bit_cast(unsigned, f);
  u += 0x7FFFu + ((u >> 16) & 1u);          // round-to-nearest-even
  return (unsigned short)(u >> 16);
}
__device__ __forceinline__ float bf2f(unsigned short s) {
  return __builtin_bit_cast(float, (unsigned)s << 16);
}

// ---------------------------------------------------------------------------
// Pack conv weights [Cout,Cin,k,k] f32 -> bf16 B-fragment tiles.
// Elem e of lane L holds B[K][N], K = 16*(e>>3) + 8*(L>>4) + (e&7), N = L&15.
// ---------------------------------------------------------------------------
__global__ void pack_weights(const float* __restrict__ w, unsigned short* __restrict__ dst,
                             int Cin, int Cout, int k, int nElems) {
  int i = blockIdx.x * 256 + threadIdx.x;
  if (i >= nElems) return;
  int nblocks = Cout >> 4, nchunks = Cin >> 5;
  int tile = i >> 9, r = i & 511;
  int L = r >> 4, e = r & 15;
  int nb = tile % nblocks;
  int cc = (tile / nblocks) % nchunks;
  int t  = tile / (nblocks * nchunks);
  int ky = t / k, kx = t % k;
  int N = L & 15, half = L >> 4;
  int K = ((e >> 3) << 4) + (half << 3) + (e & 7);
  int cout = (nb << 4) + N, cin = (cc << 5) + K;
  dst[i] = f2bf(w[((cout * Cin + cin) * k + ky) * k + kx]);
}

// ---------------------------------------------------------------------------
// f32 NCHW -> bf16 chunked [b][C/32][HW][32]
// ---------------------------------------------------------------------------
__global__ void to_chunked(const float* __restrict__ in, unsigned short* __restrict__ out,
                           int C, int nTot) {
  int i = blockIdx.x * 256 + threadIdx.x;
  if (i >= nTot) return;                     // nTot = B*(C/32)*HW*2
  int part = i & 1;
  int p    = (i >> 1) & 65535;
  int rest = i >> 17;
  int nc   = C >> 5;
  int cc   = rest % nc;
  int b    = rest / nc;
  unsigned short tmp[16];
#pragma unroll
  for (int j = 0; j < 16; ++j) {
    int ch = (cc << 5) + (part << 4) + j;
    tmp[j] = f2bf(in[(((size_t)(b * C + ch)) << 16) + p]);
  }
  unsigned short* o = out + ((((size_t)(b * nc + cc)) << 16) + p) * 32 + (part << 4);
  *(uint4*)o       = *(const uint4*)&tmp[0];
  *((uint4*)o + 1) = *(const uint4*)&tmp[8];
}

// ---------------------------------------------------------------------------
// bf16 chunked (single 32-chunk, first `C` channels valid) -> f32 NCHW
// ---------------------------------------------------------------------------
__global__ void unchunk(const unsigned short* __restrict__ in, float* __restrict__ out,
                        int C, int nTot) {
  int i = blockIdx.x * 256 + threadIdx.x;
  if (i >= nTot) return;                     // nTot = B*C*HW
  int p  = i & 65535;
  int ch = (i >> 16) % C;
  int b  = (i >> 16) / C;
  out[i] = bf2f(in[((((size_t)b) << 16) + p) * 32 + ch]);
}

// ---------------------------------------------------------------------------
// Implicit-GEMM conv via V_WMMA_F32_16X16X32_BF16, chunked bf16 in/out.
// Double-buffered LDS pipeline: async global->LDS copy of chunk cc+1 is
// issued before computing chunk cc; drained with s_wait_asynccnt + barrier.
// Grid: x = 2*H, y = batch, z = COUT/(16*NB).
// ---------------------------------------------------------------------------
template<int CIN, int COUT, int K>
__global__ __launch_bounds__(256)
void conv_wmma2(const unsigned short* __restrict__ in, unsigned short* __restrict__ out,
                const unsigned short* __restrict__ wfrag, const float* __restrict__ bias) {
  constexpr int H = 256, W = 256;
  constexpr int NCHUNK = CIN >> 5;
  constexpr int NBTOT  = COUT >> 4;
  constexpr int NB     = (COUT >= 32) ? 2 : 1;
  constexpr int OCC    = (COUT + 31) >> 5;
  constexpr int PAD    = K >> 1;
  constexpr int BUFSZ  = K * 132 * 32;            // ushorts per stage buffer

  __shared__ unsigned short s_act[2 * BUFSZ];     // double-buffered act tiles
  __shared__ unsigned short s_st[128 * 32];       // [128 px][32 ch] bf16 out tile

  const int tid  = threadIdx.x;
  const int lane = tid & 31, wave = tid >> 5;
  const int tx = blockIdx.x & 1, h = blockIdx.x >> 1;
  const int px0 = tx << 7;
  const int bb = blockIdx.y;
  const int nb0 = blockIdx.z * NB;
  const unsigned short* inb = in + (((size_t)bb * NCHUNK) << 16) * 32;

  // Issue async global->LDS copies (plus DS zero-fill for OOB halo) for chunk cc.
  auto stage = [&](int cc, int buf) {
    const int base = buf * BUFSZ;
#pragma unroll
    for (int ry = 0; ry < K; ++ry) {
      const int gy = h + ry - PAD;
      const bool yok = (gy >= 0) & (gy < H);
      for (int j = tid; j < 528; j += 256) {      // 132 px * 4 x 16B segments
        int px  = j >> 2, seg = j & 3;
        int gx  = px0 + px - 2;
        int li  = base + ((ry * 132 + px) << 5) + (seg << 3);
        if (yok & (gx >= 0) & (gx < W)) {
          const unsigned short* g =
              inb + ((((size_t)cc << 16) + gy * W + gx) << 5) + (seg << 3);
          unsigned lofs = (unsigned)(size_t)&s_act[li];
          asm volatile("global_load_async_to_lds_b128 %0, %1, off"
                       :: "v"(lofs), "v"(g) : "memory");
        } else {
          *(uint4*)&s_act[li] = make_uint4(0u, 0u, 0u, 0u);
        }
      }
    }
  };

  v8f acc0 = {}, acc1 = {};
  const int m = lane & 15, half = lane >> 4, kb = half << 3;
  const unsigned short* wlane = wfrag + ((size_t)nb0 << 9) + (lane << 4);

  stage(0, 0);
  asm volatile("s_wait_asynccnt 0" ::: "memory");
  __syncthreads();

#pragma unroll
  for (int cc = 0; cc < NCHUNK; ++cc) {
    if (cc + 1 < NCHUNK) stage(cc + 1, (cc + 1) & 1);   // prefetch next chunk

    const int base = (cc & 1) * BUFSZ;
#pragma unroll
    for (int t = 0; t < K * K; ++t) {
      int ky = t / K, kx = t % K;
      int pxl = (wave << 4) + m + kx - PAD + 2;
      int a0 = base + ((ky * 132 + pxl) << 5) + kb;
      FragBF A, B0, B1;
      A.u[0] = *(const uint4*)&s_act[a0];
      A.u[1] = *(const uint4*)&s_act[a0 + 16];
      const uint4* bp = (const uint4*)(wlane + ((size_t)((t * NCHUNK + cc) * NBTOT) << 9));
      B0.u[0] = bp[0];
      B0.u[1] = bp[1];
      acc0 = __builtin_amdgcn_wmma_f32_16x16x32_bf16(
          false, A.v, false, B0.v, (short)0, acc0, false, false);
      if constexpr (NB == 2) {
        B1.u[0] = bp[64];                   // next 512-ushort tile
        B1.u[1] = bp[65];
        acc1 = __builtin_amdgcn_wmma_f32_16x16x32_bf16(
            false, A.v, false, B1.v, (short)0, acc1, false, false);
      }
    }
    asm volatile("s_wait_asynccnt 0" ::: "memory");
    __syncthreads();
  }

  // ---- bias + ReLU -> bf16, transpose through LDS, 64B-line stores
  {
    const int n = lane & 15;
    const int cout0 = blockIdx.z * 16 * NB;
    const float bv0 = bias[cout0 + n];
    const float bv1 = (NB == 2) ? bias[cout0 + 16 + n] : 0.f;
#pragma unroll
    for (int v = 0; v < 8; ++v) {
      int pxl = (wave << 4) + v + (half << 3);
      float x0 = acc0[v] + bv0;
      s_st[(pxl << 5) + n] = f2bf(x0 > 0.f ? x0 : 0.f);
      if constexpr (NB == 2) {
        float x1 = acc1[v] + bv1;
        s_st[(pxl << 5) + 16 + n] = f2bf(x1 > 0.f ? x1 : 0.f);
      }
    }
  }
  __syncthreads();
  {
    const int px = tid >> 1, part = tid & 1;
    const uint4* sp = (const uint4*)&s_st[(px << 5) + (part << 4)];
    unsigned short* ob = out +
        ((((size_t)(bb * OCC + blockIdx.z)) << 16) + h * W + px0 + px) * 32 + (part << 4);
    uint4 d0 = sp[0], d1 = sp[1];
    *(uint4*)ob       = d0;
    *((uint4*)ob + 1) = d1;
  }
}

// ---------------------------------------------------------------------------
// Direct f32 conv for tiny-channel tail layers, fully compile-time shapes.
// ACT: 0=ReLU, 1=sigmoid.  Launched with exactly B*COUT*HW/256 blocks.
// ---------------------------------------------------------------------------
template<int CIN, int COUT, int K, int ACT>
__global__ __launch_bounds__(256)
void conv_small(const float* __restrict__ in, float* __restrict__ out,
                const float* __restrict__ w, const float* __restrict__ bias,
                int outBatchStride) {
  constexpr int H = 256, W = 256, HW = 65536;
  constexpr int PAD = K >> 1;
  int i = blockIdx.x * 256 + threadIdx.x;
  int x  = i & 255;
  int y  = (i >> 8) & 255;
  int co = (i >> 16) % COUT;
  int b  = (i >> 16) / COUT;
  float sum = bias[co];
  const float* inb = in + b * CIN * HW;
  const float* wb  = w + co * CIN * K * K;
#pragma unroll
  for (int ci = 0; ci < CIN; ++ci)
#pragma unroll
    for (int ky = 0; ky < K; ++ky) {
      int gy = y + ky - PAD;
      if (gy < 0 || gy >= H) continue;
#pragma unroll
      for (int kx = 0; kx < K; ++kx) {
        int gx = x + kx - PAD;
        if (gx < 0 || gx >= W) continue;
        sum += inb[(ci << 16) + (gy << 8) + gx] * wb[(ci * K + ky) * K + kx];
      }
    }
  if constexpr (ACT == 0) sum = sum > 0.f ? sum : 0.f;
  else                    sum = 1.f / (1.f + __expf(-sum));
  out[b * outBatchStride + co * HW + (y << 8) + x] = sum;
}

// ---------------------------------------------------------------------------
template<int CIN, int COUT, int K>
static void launch_layer(const unsigned short* in, unsigned short* out,
                         const unsigned short* wf, const float* bias, hipStream_t s) {
  constexpr int NB = (COUT >= 32) ? 2 : 1;
  dim3 g(512, 4, COUT / (16 * NB));
  conv_wmma2<CIN, COUT, K><<<g, 256, 0, s>>>(in, out, wf, bias);
}

template<int CIN, int COUT, int K, int ACT>
static void launch_small(const float* in, float* out, const float* w, const float* b,
                         int obs, hipStream_t s) {
  conv_small<CIN, COUT, K, ACT><<<4 * COUT * 65536 / 256, 256, 0, s>>>(in, out, w, b, obs);
}

extern "C" void kernel_launch(void* const* d_in, const int* in_sizes, int n_in,
                              void* d_out, int out_size, void* d_ws, size_t ws_size,
                              hipStream_t stream) {
  (void)in_sizes; (void)n_in; (void)out_size; (void)ws_size;
  const int B = 4, HW = 65536;

  const float* x = (const float*)d_in[0];
  const float* tw[9]; const float* tb[9];
  for (int i = 0; i < 9; ++i) { tw[i] = (const float*)d_in[1 + i]; tb[i] = (const float*)d_in[10 + i]; }
  const float *brw[3][10], *brb[3][10];
  for (int br = 0; br < 3; ++br)
    for (int i = 0; i < 10; ++i) {
      brw[br][i] = (const float*)d_in[19 + br * 20 + i];
      brb[br][i] = (const float*)d_in[29 + br * 20 + i];
    }

  // workspace layout
  const size_t SZC = (size_t)B * 128 * HW;            // ushorts per chunked buffer
  unsigned short* cA = (unsigned short*)d_ws;
  unsigned short* cB = cA + SZC;
  float* f0 = (float*)(cB + SZC);
  float* f1 = f0 + (size_t)B * 16 * HW;
  float* f2 = f1 + (size_t)B * 16 * HW;
  unsigned short* wf0 = (unsigned short*)(f2 + (size_t)B * 16 * HW);

  struct LW { int Cin, Cout, k; };
  const LW lw[7] = {{128,128,3},{128,128,3},{128,128,1},{128,64,3},{64,32,3},{32,32,1},{32,16,3}};
  unsigned short* wfp[7];
  size_t off = 0;
  for (int l = 0; l < 7; ++l) {
    wfp[l] = wf0 + off;
    size_t tiles = (size_t)lw[l].k * lw[l].k * (lw[l].Cin >> 5) * (lw[l].Cout >> 4);
    off += tiles * 512;
  }
  for (int l = 0; l < 7; ++l) {
    int nE = lw[l].k * lw[l].k * (lw[l].Cin >> 5) * (lw[l].Cout >> 4) * 512;
    pack_weights<<<(nE + 255) / 256, 256, 0, stream>>>(tw[l], wfp[l],
                                                       lw[l].Cin, lw[l].Cout, lw[l].k, nE);
  }

  // x -> chunked bf16
  {
    int nTot = B * (128 >> 5) * HW * 2;
    to_chunked<<<(nTot + 255) / 256, 256, 0, stream>>>(x, cA, 128, nTot);
  }

  // trunk WMMA layers (chunked bf16 ping-pong)
  launch_layer<128,128,3>(cA, cB, wfp[0], tb[0], stream);
  launch_layer<128,128,3>(cB, cA, wfp[1], tb[1], stream);
  launch_layer<128,128,1>(cA, cB, wfp[2], tb[2], stream);
  launch_layer<128, 64,3>(cB, cA, wfp[3], tb[3], stream);
  launch_layer< 64, 32,3>(cA, cB, wfp[4], tb[4], stream);
  launch_layer< 32, 32,1>(cB, cA, wfp[5], tb[5], stream);
  launch_layer< 32, 16,3>(cA, cB, wfp[6], tb[6], stream);  // 16ch chunked in cB

  // chunked bf16 -> f32 NCHW (16 ch) for the tail
  {
    int nTot = B * 16 * HW;
    unchunk<<<(nTot + 255) / 256, 256, 0, stream>>>(cB, f0, 16, nTot);
  }

  launch_small<16, 9, 3, 0>(f0, f1, tw[7], tb[7], 9 * HW, stream);
  launch_small< 9, 9, 1, 0>(f1, f0, tw[8], tb[8], 9 * HW, stream);  // h = f0 (persists)

  // branches: f1/f2 ping-pong, final sigmoid into d_out channel br
  float* outp = (float*)d_out;
  for (int br = 0; br < 3; ++br) {
    const float* w0 = brw[br][0]; (void)w0;
    launch_small<9, 9, 3, 0>(f0, f1, brw[br][0], brb[br][0], 9 * HW, stream);
    launch_small<9, 9, 3, 0>(f1, f2, brw[br][1], brb[br][1], 9 * HW, stream);
    launch_small<9, 9, 1, 0>(f2, f1, brw[br][2], brb[br][2], 9 * HW, stream);
    launch_small<9, 6, 3, 0>(f1, f2, brw[br][3], brb[br][3], 6 * HW, stream);
    launch_small<6, 6, 3, 0>(f2, f1, brw[br][4], brb[br][4], 6 * HW, stream);
    launch_small<6, 6, 1, 0>(f1, f2, brw[br][5], brb[br][5], 6 * HW, stream);
    launch_small<6, 3, 3, 0>(f2, f1, brw[br][6], brb[br][6], 3 * HW, stream);
    launch_small<3, 3, 3, 0>(f1, f2, brw[br][7], brb[br][7], 3 * HW, stream);
    launch_small<3, 3, 1, 0>(f2, f1, brw[br][8], brb[br][8], 3 * HW, stream);
    launch_small<3, 1, 3, 1>(f1, outp + (size_t)br * HW, brw[br][9], brb[br][9],
                             3 * HW, stream);
  }
}
